// DAriEL_Encoder_Cell_2_4191888081159
// MI455X (gfx1250) — matrix operand order repaired and checked by reading, not compiled, not run on hardware
//
#include <hip/hip_runtime.h>
#include <hip/hip_bf16.h>
#include <math.h>

#define VOCAB 50257
#define EDIM 64
#define HDIM 256
#define FOURH 1024
#define LATD 16
#define LSEQ 33
#define BATCH 1024
#define EPSV 1e-7f
#define SIZE_LAT 3.0f
#define NT_TILES ((VOCAB + 15) / 16)   // 3142
#define MT_TILES (BATCH / 16)          // 64

typedef __attribute__((ext_vector_type(16))) __bf16 v16bf;
typedef __attribute__((ext_vector_type(8)))  float  v8f;
typedef __attribute__((ext_vector_type(8)))  unsigned int v8u;

static __device__ __forceinline__ unsigned short f2bf(float f) {
    unsigned int u = __float_as_uint(f);
    u += 0x7FFFu + ((u >> 16) & 1u);
    return (unsigned short)(u >> 16);
}
static __device__ __forceinline__ float sigf(float x) { return 1.0f / (1.0f + expf(-x)); }

// ---------------- init: tokens_work = tokens with [:,t+1]=input_token; zero accumulators
__global__ void k_init(const int* __restrict__ tokens_in, const int* __restrict__ input_token,
                       const int* __restrict__ tstep, int* __restrict__ tokens_work,
                       float* s_all, float* s_lo, float* p_sj) {
    int i = blockIdx.x * blockDim.x + threadIdx.x;
    int t = *tstep;
    if (i < BATCH * LSEQ) {
        int b = i / LSEQ, col = i % LSEQ;
        int v = tokens_in[i];
        if (col == t + 1) v = input_token[b];
        tokens_work[i] = v;
    }
    if (i < BATCH) { s_all[i] = 0.f; s_lo[i] = 0.f; p_sj[i] = 0.f; }
}

// ---------------- pack Wd (f32 [H,V]) into WMMA B-fragment layout, bf16
// layout: uint record index = ((ntile*8 + kc)*32 + lane)*8 + v
// element v packs bf16 pair (K = kc*32 + (v<4 ? laneHi*8+2v : 16+laneHi*8+2(v-4)), +1) for column n = ntile*16+laneLo
__global__ void k_pack_b(const float* __restrict__ Wd, unsigned int* __restrict__ WdP) {
    long tid = (long)blockIdx.x * blockDim.x + threadIdx.x;   // [0, NT_TILES*8*32)
    if (tid >= (long)NT_TILES * 8 * 32) return;
    int lane = (int)(tid & 31);
    int kc   = (int)((tid >> 5) & 7);
    int nt   = (int)(tid >> 8);
    int laneLo = lane & 15, laneHi = (lane >> 4) & 1;
    int n = nt * 16 + laneLo;
    int nc = min(n, VOCAB - 1);
    bool valid = (n < VOCAB);
    v8u rec;
#pragma unroll
    for (int v = 0; v < 8; ++v) {
        int k = kc * 32 + ((v < 4) ? (laneHi * 8 + 2 * v) : (16 + laneHi * 8 + 2 * (v - 4)));
        unsigned int lo = (unsigned int)f2bf(Wd[(long)k * VOCAB + nc]);
        unsigned int hi = (unsigned int)f2bf(Wd[(long)(k + 1) * VOCAB + nc]);
        rec[v] = valid ? (lo | (hi << 16)) : 0u;
    }
    *(v8u*)(WdP + tid * 8) = rec;
}

// ---------------- pack h (bf16 [B,H]) into WMMA A-fragment layout
// element v packs bf16 pair for row m = mtile*16+laneLo at
// K = kc*32 + (v<4 ? laneHi*8+2v : 16+laneHi*8+2(v-4))
__global__ void k_pack_a(const unsigned short* __restrict__ hbf, unsigned int* __restrict__ APk) {
    int tid = blockIdx.x * blockDim.x + threadIdx.x;          // [0, MT_TILES*8*32)
    if (tid >= MT_TILES * 8 * 32) return;
    int lane = tid & 31;
    int kc   = (tid >> 5) & 7;
    int mt   = tid >> 8;
    int laneLo = lane & 15, laneHi = (lane >> 4) & 1;
    const unsigned short* row = hbf + (long)(mt * 16 + laneLo) * HDIM;
    v8u rec;
#pragma unroll
    for (int v = 0; v < 8; ++v) {
        int kb = kc * 32 + ((v < 4) ? (laneHi * 8 + 2 * v) : (16 + laneHi * 8 + 2 * (v - 4)));
        rec[v] = *(const unsigned int*)(row + kb);
    }
    *(v8u*)(APk + (long)tid * 8) = rec;
}

// ---------------- LSTM forward over T = t+1 steps; one block (256 thr) per batch row
__global__ void k_lstm(const int* __restrict__ tokens_work, const float* __restrict__ emb,
                       const float* __restrict__ Wx, const float* __restrict__ Wh,
                       const float* __restrict__ b_lstm, const int* __restrict__ tstep,
                       unsigned short* __restrict__ hbf_out) {
    __shared__ float xs[EDIM];
    __shared__ float hs[HDIM];
    int b = blockIdx.x;
    int j = threadIdx.x;
    int T = *tstep + 1;
    float c = 0.f;
    hs[j] = 0.f;
    __syncthreads();
    for (int s = 0; s < T; ++s) {
        int tok = tokens_work[b * LSEQ + s];
        if (j < EDIM) xs[j] = emb[(long)tok * EDIM + j];
        __syncthreads();
        float zi = b_lstm[j], zf = b_lstm[j + HDIM], zg = b_lstm[j + 2 * HDIM], zo = b_lstm[j + 3 * HDIM];
        for (int e = 0; e < EDIM; ++e) {
            float x = xs[e];
            const float* w = Wx + (long)e * FOURH;
            zi += x * w[j]; zf += x * w[j + HDIM]; zg += x * w[j + 2 * HDIM]; zo += x * w[j + 3 * HDIM];
        }
        if (s > 0) {
            for (int k = 0; k < HDIM; ++k) {
                float h = hs[k];
                const float* w = Wh + (long)k * FOURH;
                zi += h * w[j]; zf += h * w[j + HDIM]; zg += h * w[j + 2 * HDIM]; zo += h * w[j + 3 * HDIM];
            }
        }
        c = sigf(zf) * c + sigf(zi) * tanhf(zg);
        float hn = sigf(zo) * tanhf(c);
        __syncthreads();
        hs[j] = hn;
        __syncthreads();
    }
    hbf_out[b * HDIM + j] = f2bf(hs[j]);
}

// ---------------- pad-token LM hidden state (single step, token 0), one block 256 thr
__global__ void k_pad_lstm(const float* __restrict__ emb, const float* __restrict__ Wx,
                           const float* __restrict__ b_lstm, float* __restrict__ h_pad) {
    __shared__ float xs[EDIM];
    int j = threadIdx.x;
    if (j < EDIM) xs[j] = emb[j];  // token PAD == 0
    __syncthreads();
    float zi = b_lstm[j], zf = b_lstm[j + HDIM], zg = b_lstm[j + 2 * HDIM], zo = b_lstm[j + 3 * HDIM];
    for (int e = 0; e < EDIM; ++e) {
        float x = xs[e];
        const float* w = Wx + (long)e * FOURH;
        zi += x * w[j]; zf += x * w[j + HDIM]; zg += x * w[j + 2 * HDIM]; zo += x * w[j + 3 * HDIM];
    }
    (void)zf;                                // c0 = 0, forget term vanishes
    float c = sigf(zi) * tanhf(zg);
    h_pad[j] = sigf(zo) * tanhf(c);
}

// ---------------- pad logits: logits_pad[j] = h_pad . Wd[:,j] + bd[j]
__global__ void k_logits_pad(const float* __restrict__ h_pad, const float* __restrict__ Wd,
                             const float* __restrict__ bd, float* __restrict__ logits_pad) {
    __shared__ float hsh[HDIM];
    int j0 = blockIdx.x * blockDim.x + threadIdx.x;
    if (threadIdx.x < HDIM) hsh[threadIdx.x] = h_pad[threadIdx.x];
    __syncthreads();
    if (j0 < VOCAB) {
        float acc = bd[j0];
        for (int k = 0; k < HDIM; ++k) acc += hsh[k] * Wd[(long)k * VOCAB + j0];
        logits_pad[j0] = acc;
    }
}

// ---------------- pad softmax + inclusive cumsum (single block, 1024 thr)
__global__ void k_pad_scan(const float* __restrict__ logits_pad, float* __restrict__ cum_pad,
                           float* __restrict__ p_pad) {
    __shared__ float red[1024];
    __shared__ float scan[1024];
    __shared__ float carry_sh;
    int j = threadIdx.x;
    float m = -1e30f;
    for (int i = j; i < VOCAB; i += 1024) m = fmaxf(m, logits_pad[i]);
    red[j] = m; __syncthreads();
    for (int s = 512; s > 0; s >>= 1) { if (j < s) red[j] = fmaxf(red[j], red[j + s]); __syncthreads(); }
    float rmax = red[0]; __syncthreads();
    float sm = 0.f;
    for (int i = j; i < VOCAB; i += 1024) sm += expf(logits_pad[i] - rmax);
    red[j] = sm; __syncthreads();
    for (int s = 512; s > 0; s >>= 1) { if (j < s) red[j] += red[j + s]; __syncthreads(); }
    float inv = 1.0f / red[0]; __syncthreads();
    if (j == 0) carry_sh = 0.f;
    __syncthreads();
    int nchunks = (VOCAB + 1023) / 1024;
    for (int ck = 0; ck < nchunks; ++ck) {
        int i = ck * 1024 + j;
        float v = (i < VOCAB) ? expf(logits_pad[i] - rmax) * inv : 0.f;
        scan[j] = v; __syncthreads();
        for (int off = 1; off < 1024; off <<= 1) {
            float add = (j >= off) ? scan[j - off] : 0.f;
            __syncthreads();
            scan[j] += add;
            __syncthreads();
        }
        float carry = carry_sh;
        if (i < VOCAB) { cum_pad[i] = carry + scan[j]; p_pad[i] = v; }
        __syncthreads();
        if (j == 1023) carry_sh = carry + scan[1023];
        __syncthreads();
    }
}

// ---- WMMA GEMM tile from packed fragments: 8 chunks of K=32, all-b128 loads ----
static __device__ __forceinline__ v8f gemm_tile(const v8u* __restrict__ APv,
                                                const v8u* __restrict__ BPv,
                                                int mtile, int ntile, int lane) {
    v8f acc = {};
#pragma unroll
    for (int kc = 0; kc < 8; ++kc) {
        v8u araw = APv[((long)mtile * 8 + kc) * 32 + lane];
        v8u braw = BPv[((long)ntile * 8 + kc) * 32 + lane];
        v16bf a = __builtin_bit_cast(v16bf, araw);
        v16bf b = __builtin_bit_cast(v16bf, braw);
        acc = __builtin_amdgcn_wmma_f32_16x16x32_bf16(false, a, false, b, (short)0, acc, false, false);
    }
    return acc;
}

// ---------------- fused GEMM + softmax-CDF row reduction (soft_ctx path)
__global__ void k_gemm_reduce(const unsigned int* __restrict__ APk,
                              const unsigned int* __restrict__ WdP,
                              const float* __restrict__ bd, const int* __restrict__ s_j,
                              float* s_all, float* s_lo, float* p_sj) {
    const int NSPLIT = 64;
    int wave = (blockIdx.x * blockDim.x + threadIdx.x) >> 5;
    int lane = threadIdx.x & 31;
    int mtile = wave / NSPLIT;
    int nsplit = wave % NSPLIT;
    if (mtile >= MT_TILES) return;
    int laneLo = lane & 15, laneHi = (lane >> 4) & 1;
    int m0 = mtile * 16;
    const v8u* APv = (const v8u*)APk;
    const v8u* BPv = (const v8u*)WdP;

    int sj[8];
#pragma unroll
    for (int v = 0; v < 8; ++v) sj[v] = s_j[m0 + laneHi * 8 + v];

    float rs_all[8], rs_lo[8];
#pragma unroll
    for (int v = 0; v < 8; ++v) { rs_all[v] = 0.f; rs_lo[v] = 0.f; }

    for (int nt = nsplit; nt < NT_TILES; nt += NSPLIT) {
        int n = nt * 16 + laneLo;
        bool valid = (n < VOCAB);
        v8f acc = gemm_tile(APv, BPv, mtile, nt, lane);
        if (valid) {
            float add = bd[n];
#pragma unroll
            for (int v = 0; v < 8; ++v) {
                float ev = expf(acc[v] + add);   // logits are O(1): exp without max is safe
                rs_all[v] += ev;
                if (n <= sj[v]) rs_lo[v] += ev;
                if (n == sj[v]) atomicAdd(&p_sj[m0 + laneHi * 8 + v], ev);
            }
        }
    }
#pragma unroll
    for (int v = 0; v < 8; ++v) {
        float a1 = rs_all[v], l1 = rs_lo[v];
        for (int off = 1; off < 16; off <<= 1) {
            a1 += __shfl_xor(a1, off, 32);
            l1 += __shfl_xor(l1, off, 32);
        }
        if (laneLo == 0) {
            atomicAdd(&s_all[m0 + laneHi * 8 + v], a1);
            atomicAdd(&s_lo[m0 + laneHi * 8 + v], l1);
        }
    }
}

// ---------------- CDF inversion, token/unfold outputs
__global__ void k_search(const int* __restrict__ tstep, const int* __restrict__ curdim,
                         const float* __restrict__ s_all, const float* __restrict__ s_lo,
                         const float* __restrict__ p_sj, const float* __restrict__ cum_pad,
                         const float* __restrict__ p_pad, const float* __restrict__ unfolding_point,
                         const float* __restrict__ one_softmax_in, int* __restrict__ tokens_work,
                         float* __restrict__ out_tokens, float* __restrict__ out_unfold) {
    int b = blockIdx.x * blockDim.x + threadIdx.x;
    if (b >= BATCH) return;
    int t = *tstep, d = *curdim;
    float xn;
    if (t > 0) {
        float inv = 1.0f / s_all[b];
        float chi0 = s_lo[b] * inv;
        float clo0 = chi0 - p_sj[b] * inv;   // = c_lo; new_low = 3*c_lo, xn = new_low/3
        xn = fminf(fmaxf(clo0, 0.f), 1.f - EPSV);
    } else {
        xn = fminf(fmaxf(unfolding_point[b * LATD + d] / SIZE_LAT, 0.f), 1.f - EPSV);
    }
    int idx; float chi, clo;
    if (t > 0) {
        int lo = 0, hi = VOCAB;
        while (lo < hi) { int mid = (lo + hi) >> 1; if (cum_pad[mid] < xn) lo = mid + 1; else hi = mid; }
        idx = min(lo, VOCAB - 1);
        chi = cum_pad[idx];
        clo = chi - p_pad[idx];
    } else {
        const float* row = one_softmax_in + (long)b * VOCAB;
        float c = 0.f, pv = 0.f; int cnt = 0;
        for (int jj = 0; jj < VOCAB; ++jj) {
            pv = row[jj]; c += pv;
            if (c < xn) cnt++; else break;
        }
        idx = min(cnt, VOCAB - 1);
        chi = c; clo = c - pv;
    }
    float x_new = (xn - clo) / (chi - clo + EPSV);
    tokens_work[b * LSEQ + t] = idx;
    for (int i = 0; i < LSEQ; ++i) out_tokens[b * LSEQ + i] = (float)tokens_work[b * LSEQ + i];
    for (int i = 0; i < LATD; ++i) {
        float uv = (t > 0) ? 0.f : unfolding_point[b * LATD + i];
        out_unfold[b * LATD + i] = (i == d) ? x_new : uv;
    }
}

// ---------------- WMMA GEMM -> logits materialized in d_out
__global__ void k_gemm_store(const unsigned int* __restrict__ APk,
                             const unsigned int* __restrict__ WdP,
                             const float* __restrict__ bd, float* __restrict__ out) {
    long wave = ((long)blockIdx.x * blockDim.x + threadIdx.x) >> 5;
    int lane = threadIdx.x & 31;
    long total = (long)MT_TILES * NT_TILES;
    if (wave >= total) return;
    int mtile = (int)(wave / NT_TILES);
    int ntile = (int)(wave % NT_TILES);
    int laneLo = lane & 15, laneHi = (lane >> 4) & 1;
    int m0 = mtile * 16;

    v8f acc = gemm_tile((const v8u*)APk, (const v8u*)WdP, mtile, ntile, lane);

    int n = ntile * 16 + laneLo;
    if (n < VOCAB) {
        float add = bd[n];
#pragma unroll
        for (int v = 0; v < 8; ++v)
            out[(long)(m0 + laneHi * 8 + v) * VOCAB + n] = acc[v] + add;
    }
}

// ---------------- per-row softmax normalization in place (one block per row)
__global__ void k_softmax(float* __restrict__ out) {
    __shared__ float red[256];
    float* row = out + (long)blockIdx.x * VOCAB;
    int j = threadIdx.x;
    float m = -1e30f;
    for (int i = j; i < VOCAB; i += 256) m = fmaxf(m, row[i]);
    red[j] = m; __syncthreads();
    for (int s = 128; s > 0; s >>= 1) { if (j < s) red[j] = fmaxf(red[j], red[j + s]); __syncthreads(); }
    m = red[0]; __syncthreads();
    float sm = 0.f;
    for (int i = j; i < VOCAB; i += 256) sm += expf(row[i] - m);
    red[j] = sm; __syncthreads();
    for (int s = 128; s > 0; s >>= 1) { if (j < s) red[j] += red[j + s]; __syncthreads(); }
    float inv = 1.0f / red[0];
    for (int i = j; i < VOCAB; i += 256) row[i] = expf(row[i] - m) * inv;
}

static inline char* wsal(char*& p, size_t bytes) {
    char* r = p;
    p += (bytes + 255) & ~(size_t)255;
    return r;
}

extern "C" void kernel_launch(void* const* d_in, const int* in_sizes, int n_in,
                              void* d_out, int out_size, void* d_ws, size_t ws_size,
                              hipStream_t stream) {
    const int*   input_token = (const int*)d_in[0];
    const float* one_softmax = (const float*)d_in[1];
    const int*   tokens      = (const int*)d_in[2];
    const float* unfolding   = (const float*)d_in[3];
    const float* emb         = (const float*)d_in[4];
    const float* Wx          = (const float*)d_in[5];
    const float* Wh          = (const float*)d_in[6];
    const float* b_lstm      = (const float*)d_in[7];
    const float* Wd          = (const float*)d_in[8];
    const float* bd          = (const float*)d_in[9];
    const int*   curDim      = (const int*)d_in[10];
    const int*   timeStep    = (const int*)d_in[11];

    float* out_softmax = (float*)d_out;                       // [B,V]
    float* out_tokens  = out_softmax + (long)BATCH * VOCAB;   // [B,L] as float
    float* out_unfold  = out_tokens + (long)BATCH * LSEQ;     // [B,LAT]

    char* p = (char*)d_ws;
    int*            tokens_work = (int*)wsal(p, (size_t)BATCH * LSEQ * 4);
    float*          s_all       = (float*)wsal(p, BATCH * 4);
    float*          s_lo        = (float*)wsal(p, BATCH * 4);
    float*          p_sj        = (float*)wsal(p, BATCH * 4);
    unsigned short* hbf_ctx     = (unsigned short*)wsal(p, (size_t)BATCH * HDIM * 2);
    unsigned short* hbf_new     = (unsigned short*)wsal(p, (size_t)BATCH * HDIM * 2);
    unsigned int*   APk_ctx     = (unsigned int*)wsal(p, (size_t)MT_TILES * 8 * 32 * 8 * 4);
    unsigned int*   APk_new     = (unsigned int*)wsal(p, (size_t)MT_TILES * 8 * 32 * 8 * 4);
    float*          h_pad       = (float*)wsal(p, HDIM * 4);
    float*          logits_pad  = (float*)wsal(p, (size_t)VOCAB * 4);
    float*          cum_pad     = (float*)wsal(p, (size_t)VOCAB * 4);
    float*          p_pad       = (float*)wsal(p, (size_t)VOCAB * 4);
    unsigned int*   WdP         = (unsigned int*)wsal(p, (size_t)NT_TILES * 8 * 32 * 8 * 4);
    (void)ws_size; (void)n_in; (void)in_sizes; (void)out_size;

    // 1. tokens_work + accumulator zeroing
    k_init<<<(BATCH * LSEQ + 255) / 256, 256, 0, stream>>>(tokens, input_token, timeStep,
                                                           tokens_work, s_all, s_lo, p_sj);
    // 2. Wd -> packed bf16 B fragments
    k_pack_b<<<(NT_TILES * 8 * 32 + 255) / 256, 256, 0, stream>>>(Wd, WdP);
    // 3. LSTM over tokens[:, :t+1] (original col t), then pack A fragments
    k_lstm<<<BATCH, 256, 0, stream>>>(tokens_work, emb, Wx, Wh, b_lstm, timeStep, hbf_ctx);
    k_pack_a<<<(MT_TILES * 8 * 32 + 255) / 256, 256, 0, stream>>>(hbf_ctx, APk_ctx);
    // 4-6. pad-token LM: hidden, logits, softmax+cumsum (batch-degenerate: one row)
    k_pad_lstm<<<1, 256, 0, stream>>>(emb, Wx, b_lstm, h_pad);
    k_logits_pad<<<(VOCAB + 255) / 256, 256, 0, stream>>>(h_pad, Wd, bd, logits_pad);
    k_pad_scan<<<1, 1024, 0, stream>>>(logits_pad, cum_pad, p_pad);
    // 7. fused WMMA GEMM + CDF reductions (c_hi/c_lo ingredients)
    k_gemm_reduce<<<(MT_TILES * 64) / 8, 256, 0, stream>>>(APk_ctx, WdP, bd, input_token,
                                                           s_all, s_lo, p_sj);
    // 8. CDF inversion -> idx, tokens[:,t], unfold/tokens outputs
    k_search<<<(BATCH + 255) / 256, 256, 0, stream>>>(timeStep, curDim, s_all, s_lo, p_sj,
                                                      cum_pad, p_pad, unfolding, one_softmax,
                                                      tokens_work, out_tokens, out_unfold);
    // 9. LSTM again with updated col t, pack A
    k_lstm<<<BATCH, 256, 0, stream>>>(tokens_work, emb, Wx, Wh, b_lstm, timeStep, hbf_new);
    k_pack_a<<<(MT_TILES * 8 * 32 + 255) / 256, 256, 0, stream>>>(hbf_new, APk_new);
    // 10. WMMA GEMM: logits -> d_out
    k_gemm_store<<<(int)(((long)MT_TILES * NT_TILES) / 8), 256, 0, stream>>>(APk_new, WdP,
                                                                             bd, out_softmax);
    // 11. softmax normalize in place
    k_softmax<<<BATCH, 256, 0, stream>>>(out_softmax);
}
